// Model_43095701848143
// MI455X (gfx1250) — matrix operand order repaired and checked by reading, not compiled, not run on hardware
//
#include <hip/hip_runtime.h>
#include <hip/hip_bf16.h>

// ---------------------------------------------------------------------------
// biLSTM + synonym attention for MI455X (gfx1250, wave32, WMMA f16->f32)
//
// Pipeline:
//   1. pack weights -> f16 transposed [Npad][Kpad], gate-permuted for LSTM
//   2. gather embeddings -> f16 [8192][320]
//   3. WMMA GEMM: xk = emb @ Wk + b (gate-permuted cols)   (both directions)
//   4. persistent LSTM scan kernel (1 WG/direction): Wr staged into LDS via
//      async-copy (ASYNCcnt), h in LDS, c in regs, per-step WMMA,
//      gates via 4-lane shfl exchange
//   5. WMMA GEMM: out = hidden @ W1 + b1
//   6. attention (gathers + exp/tanh) + weighted sum over sequence
// ---------------------------------------------------------------------------

typedef _Float16 v8h  __attribute__((ext_vector_type(8)));
typedef _Float16 v16h __attribute__((ext_vector_type(16)));
typedef float    v8f  __attribute__((ext_vector_type(8)));

#define BB   64
#define SS   128
#define BS   8192        // B*S
#define DD   300
#define DP   320         // D padded to mult of 32 (K of Wk / W1)
#define HH   150
#define HP   160         // H padded (K of Wr)
#define GG   600         // 4H
#define GP   608         // 4H padded to mult of 16 (N of Wk/Wr)
#define NTG  38          // GP/16 N-tiles for gate GEMMs
#define NP1  304         // W1 N padded
#define LDH  304         // hidden row stride (f32)

__device__ __forceinline__ v8f wmma_f16(v16h a, v16h b, v8f c) {
  // D = A(16x32 f16) x B(32x16 f16) + C(16x16 f32)
  return __builtin_amdgcn_wmma_f32_16x16x32_f16(
      /*neg_a=*/false, a, /*neg_b=*/false, b,
      /*c_mod=*/(short)0, c, /*reuse_a=*/false, /*reuse_b=*/false);
}

__device__ __forceinline__ v16h load_a_frag(const _Float16* __restrict__ base) {
  // lane layout (16-bit A 16x32): lanes0-15 K=0..7 & 16..23, lanes16-31 +8
  v8h lo = *(const v8h*)(base);
  v8h hi = *(const v8h*)(base + 16);
  return __builtin_shufflevector(lo, hi, 0,1,2,3,4,5,6,7,8,9,10,11,12,13,14,15);
}

__device__ __forceinline__ float sigm(float x) { return 1.f / (1.f + expf(-x)); }

// LDS byte-offset of a generic pointer known to point into LDS.
__device__ __forceinline__ unsigned lds_off_of(const void* p) {
  return (unsigned)(unsigned long long)
      (__attribute__((address_space(3))) const void*)p;
}

// Async copy 16B global -> LDS (CDNA5 GLOBAL_LOAD_ASYNC_TO_LDS_B128, ASYNCcnt)
__device__ __forceinline__ void async_g2l_b128(unsigned lds_byte_off,
                                               const void* gptr) {
  asm volatile("global_load_async_to_lds_b128 %0, %1, off"
               :
               : "v"(lds_byte_off), "v"((unsigned long long)gptr)
               : "memory");
}

__device__ __forceinline__ void wait_asynccnt0() {
#if __has_builtin(__builtin_amdgcn_s_wait_asynccnt)
  __builtin_amdgcn_s_wait_asynccnt(0);
#else
  asm volatile("s_wait_asynccnt 0" ::: "memory");
#endif
}

// ---------------------------------------------------------------------------
// Packing kernels
// ---------------------------------------------------------------------------

// W[K][N] f32 -> Wt[Np][Kp] f16 (transposed, zero padded).
// perm!=0: output column n' maps to original column (n'&3)*150 + (n'>>2)
// so the 4 gates of unit j occupy 4 adjacent packed columns.
__global__ void pack_weight_t(const float* __restrict__ W, int K, int N,
                              _Float16* __restrict__ Wt, int Kp, int Np, int perm) {
  int idx = blockIdx.x * blockDim.x + threadIdx.x;
  if (idx >= Np * Kp) return;
  int n = idx / Kp, k = idx % Kp;
  bool ok; int on;
  if (perm) { int j = n >> 2, g = n & 3; ok = (k < K) && (j < HH); on = g * HH + j; }
  else      { ok = (k < K) && (n < N);   on = n; }
  Wt[idx] = (_Float16)(ok ? W[k * N + on] : 0.f);
}

__global__ void pack_bias_perm(const float* __restrict__ b, float* __restrict__ bp) {
  int n = blockIdx.x * blockDim.x + threadIdx.x;
  if (n >= GP) return;
  int j = n >> 2, g = n & 3;
  bp[n] = (j < HH) ? b[g * HH + j] : 0.f;
}

__global__ void gather_pack_emb(const int* __restrict__ tokens,
                                const float* __restrict__ embm,
                                _Float16* __restrict__ out) {
  int idx = blockIdx.x * blockDim.x + threadIdx.x;
  if (idx >= BS * DP) return;
  int row = idx / DP, c = idx % DP;
  int tok = tokens[row];
  float v = (c < DD) ? embm[tok * DD + c] : 0.f;
  out[idx] = (_Float16)v;
}

__global__ void pack_rows_f16(const float* __restrict__ X, int ldx, int C,
                              _Float16* __restrict__ Y, int Cp, int total) {
  int idx = blockIdx.x * blockDim.x + threadIdx.x;
  if (idx >= total) return;
  int row = idx / Cp, c = idx % Cp;
  Y[idx] = (_Float16)((c < C) ? X[row * ldx + c] : 0.f);
}

// ---------------------------------------------------------------------------
// Generic WMMA GEMM: C[M][ldc] = A[M][Kp](f16) * Bt[Np][Kp](f16,transposed) + bias
// One 16x16 tile per wave.
// ---------------------------------------------------------------------------
__global__ __launch_bounds__(256) void wmma_gemm_f16(
    const _Float16* __restrict__ A, const _Float16* __restrict__ Bt,
    const float* __restrict__ bias, int biasN,
    float* __restrict__ C, int Kp, int ldc, int ntN, int totalTiles) {
  int wave = blockIdx.x * (blockDim.x >> 5) + (threadIdx.x >> 5);
  int lane = threadIdx.x & 31;
  if (wave >= totalTiles) return;
  int mt = wave / ntN, nt = wave % ntN;
  int m0 = mt * 16, n0 = nt * 16;
  int half = lane >> 4, l15 = lane & 15;

  const _Float16* aPtr = A + (m0 + l15) * Kp + half * 8;
  const _Float16* bPtr = Bt + (n0 + l15) * Kp + half * 8;

  v8f acc = {0.f, 0.f, 0.f, 0.f, 0.f, 0.f, 0.f, 0.f};
  for (int k0 = 0; k0 < Kp; k0 += 32) {
    __builtin_prefetch(bPtr + k0 + 64, 0, 0);       // global_prefetch_b8
    v16h a = load_a_frag(aPtr + k0);
    v16h b = load_a_frag(bPtr + k0);                // lanes0-15: K=0..7 & 16..23 @N=lane
    acc = wmma_f16(a, b, acc);
  }
  int col = n0 + l15;
  float bv = (bias && col < biasN) ? bias[col] : 0.f;
#pragma unroll
  for (int r = 0; r < 8; ++r) {
    int row = m0 + r + half * 8;                    // f32 C layout: VGPR r -> M
    C[(long)row * ldc + col] = acc[r] + bv;
  }
}

// ---------------------------------------------------------------------------
// Persistent LSTM scan: grid=2 (dir), block=1024 (32 waves).
// Dynamic LDS: WrT (f16 [608][160], 194.6 KB, async-staged once) + h (20.5 KB).
// c lives in registers (static tile->wave map). Gate columns are permuted
// (col' = j*4 + gate) so i/f/g/o for unit j sit in 4 adjacent lanes of one
// accumulator tile -> exchange via __shfl.
// ---------------------------------------------------------------------------
#define TOTT (4 * NTG)        // 152 tiles: 4 M-tiles x 38 N-tiles
#define TPW  5                // ceil(152/32) tiles per wave
#define WR_BYTES   ((size_t)GP * HP * 2)          // 194560
#define H_BYTES    ((size_t)BB * HP * 2)          // 20480
#define LSTM_SMEM  (WR_BYTES + H_BYTES)           // 215040 (< 320KB WGP LDS)

__global__ __launch_bounds__(1024) void lstm_scan(
    const float* __restrict__ xkF, const float* __restrict__ xkB,
    const _Float16* __restrict__ WrTF, const _Float16* __restrict__ WrTB,
    float* __restrict__ hidden) {
  extern __shared__ char lsmem[];
  _Float16* wr  = (_Float16*)lsmem;              // [GP][HP]
  _Float16* hsh = (_Float16*)(lsmem + WR_BYTES); // [BB][HP]

  int dir = blockIdx.x;
  const float*    xk  = dir ? xkB  : xkF;
  const _Float16* WrT = dir ? WrTB : WrTF;

  int tid = threadIdx.x;
  int wave = tid >> 5, lane = tid & 31;
  int half = lane >> 4, l15 = lane & 15;

  // ---- stage Wr into LDS with CDNA5 async global->LDS copies ----
  {
    unsigned wr_off = lds_off_of(wr);
    const int CHUNKS = (int)(WR_BYTES / 16);     // 12160 x 16B
    for (int i = tid; i < CHUNKS; i += 1024) {
      async_g2l_b128(wr_off + (unsigned)i * 16u, (const void*)(WrT + i * 8));
    }
  }
  for (int i = tid; i < BB * HP; i += 1024) hsh[i] = (_Float16)0.f;

  float cst[TPW][8];
#pragma unroll
  for (int i = 0; i < TPW; ++i)
#pragma unroll
    for (int r = 0; r < 8; ++r) cst[i][r] = 0.f;

  wait_asynccnt0();     // drain this wave's async copies
  __syncthreads();      // all waves' copies + h init visible

  for (int t = 0; t < SS; ++t) {
    int pos = dir ? (SS - 1 - t) : t;
    v8f acc[TPW];

    // ---- GEMM phase: z = h @ Wr + xk[:, pos, :]  (A and B from LDS) ----
#pragma unroll
    for (int i = 0; i < TPW; ++i) {
      int tile = wave + i * 32;
      if (tile < TOTT) {
        int mt = tile & 3, nt = tile >> 2;
        int m0 = mt * 16, n0 = nt * 16;
        const _Float16* aPtr = hsh + (m0 + l15) * HP + half * 8;
        const _Float16* bPtr = wr  + (n0 + l15) * HP + half * 8;
        v8f a8 = {0.f, 0.f, 0.f, 0.f, 0.f, 0.f, 0.f, 0.f};
#pragma unroll
        for (int k0 = 0; k0 < HP; k0 += 32) {
          v16h a = load_a_frag(aPtr + k0);
          v16h b = load_a_frag(bPtr + k0);
          a8 = wmma_f16(a, b, a8);
        }
        int col = n0 + l15;
#pragma unroll
        for (int r = 0; r < 8; ++r) {
          int bb = m0 + r + half * 8;
          a8[r] += xk[((long)bb * SS + pos) * GP + col];
        }
        acc[i] = a8;
      }
    }
    __syncthreads();   // all reads of hsh complete before it is rewritten

    // ---- Gate phase: exchange i/f/g/o across the 4-lane group, update c,h ----
#pragma unroll
    for (int i = 0; i < TPW; ++i) {
      int tile = wave + i * 32;
      if (tile < TOTT) {
        int mt = tile & 3, nt = tile >> 2;
        int m0 = mt * 16, n0 = nt * 16;
        int col = n0 + l15;
        int j = col >> 2;
        int g0 = lane & ~3;
#pragma unroll
        for (int r = 0; r < 8; ++r) {
          float v  = acc[i][r];
          float zi = __shfl(v, g0 + 0, 32);
          float zf = __shfl(v, g0 + 1, 32);
          float zg = __shfl(v, g0 + 2, 32);
          float zo = __shfl(v, g0 + 3, 32);
          float cv = sigm(zf) * cst[i][r] + sigm(zi) * tanhf(zg);
          float hv = sigm(zo) * tanhf(cv);
          cst[i][r] = cv;
          if (((lane & 3) == 0) && col < GG) {
            int bb = m0 + r + half * 8;
            hsh[bb * HP + j] = (_Float16)hv;
            hidden[((long)bb * SS + pos) * LDH + dir * HH + j] = hv;
          }
        }
      }
    }
    __syncthreads();   // hsh writes visible before next step's GEMM
  }
}

// ---------------------------------------------------------------------------
// Attention: one wave per token.
// ---------------------------------------------------------------------------
__global__ __launch_bounds__(256) void attn_kernel(
    const int* __restrict__ tokens, const float* __restrict__ embm,
    const int* __restrict__ synIdx, const float* __restrict__ hidden,
    const float* __restrict__ outw1, const float* __restrict__ W2,
    const float* __restrict__ b2, float* __restrict__ hhat,
    float* __restrict__ c2buf) {
  int wave = blockIdx.x * (blockDim.x >> 5) + (threadIdx.x >> 5);
  int lane = threadIdx.x & 31;
  if (wave >= BS) return;
  int tok = tokens[wave];
  const float* e0 = embm + (long)synIdx[tok * 4 + 0] * DD;
  const float* e1 = embm + (long)synIdx[tok * 4 + 1] * DD;
  const float* e2 = embm + (long)synIdx[tok * 4 + 2] * DD;
  const float* e3 = embm + (long)synIdx[tok * 4 + 3] * DD;
  const float* o  = outw1 + (long)wave * NP1;

  float d0 = 0.f, d1 = 0.f, d2 = 0.f, d3 = 0.f;
  for (int d = lane; d < DD; d += 32) {
    float od = o[d];
    d0 += e0[d] * od; d1 += e1[d] * od; d2 += e2[d] * od; d3 += e3[d] * od;
  }
#pragma unroll
  for (int m = 16; m; m >>= 1) {
    d0 += __shfl_xor(d0, m, 32); d1 += __shfl_xor(d1, m, 32);
    d2 += __shfl_xor(d2, m, 32); d3 += __shfl_xor(d3, m, 32);
  }
  float cf0 = expf(d0), cf1 = expf(d1), cf2 = expf(d2), cf3 = expf(d3);

  const float* hid = hidden + (long)wave * LDH;
  float w2a = 0.f;
  for (int d = lane; d < DD; d += 32) {
    float hh = cf0 * e0[d] + cf1 * e1[d] + cf2 * e2[d] + cf3 * e3[d] + hid[d];
    hhat[(long)wave * LDH + d] = hh;
    w2a += hh * W2[d];
  }
#pragma unroll
  for (int m = 16; m; m >>= 1) w2a += __shfl_xor(w2a, m, 32);
  if (lane == 0) c2buf[wave] = expf(tanhf(w2a + b2[0]));
}

__global__ void reduce_kernel(const float* __restrict__ hhat,
                              const float* __restrict__ c2buf,
                              float* __restrict__ out) {
  int b = blockIdx.x;
  for (int d = threadIdx.x; d < DD; d += blockDim.x) {
    float acc = 0.f;
    for (int s = 0; s < SS; ++s) {
      int row = b * SS + s;
      acc += c2buf[row] * hhat[(long)row * LDH + d];
    }
    out[b * DD + d] = acc;
  }
}

// ---------------------------------------------------------------------------
// Host-side launcher
// ---------------------------------------------------------------------------
extern "C" void kernel_launch(void* const* d_in, const int* in_sizes, int n_in,
                              void* d_out, int out_size, void* d_ws, size_t ws_size,
                              hipStream_t stream) {
  const int*   batch_inputs = (const int*)d_in[0];
  const float* embm         = (const float*)d_in[1];
  const int*   syn_idx      = (const int*)d_in[2];
  const float* Wk_f         = (const float*)d_in[3];
  const float* Wr_f         = (const float*)d_in[4];
  const float* b_f          = (const float*)d_in[5];
  const float* Wk_b         = (const float*)d_in[6];
  const float* Wr_b         = (const float*)d_in[7];
  const float* b_b          = (const float*)d_in[8];
  const float* W1           = (const float*)d_in[9];
  const float* b1           = (const float*)d_in[10];
  const float* W2           = (const float*)d_in[11];
  const float* b2           = (const float*)d_in[12];
  float* out = (float*)d_out;

  // workspace carving (256B aligned)
  char* p = (char*)d_ws;
  auto carve = [&](size_t bytes) -> void* {
    void* r = (void*)p;
    p += (bytes + 255) & ~((size_t)255);
    return r;
  };
  _Float16* emb_h = (_Float16*)carve((size_t)BS * DP * 2);
  _Float16* wkf_t = (_Float16*)carve((size_t)GP * DP * 2);
  _Float16* wkb_t = (_Float16*)carve((size_t)GP * DP * 2);
  _Float16* wrf_t = (_Float16*)carve((size_t)GP * HP * 2);
  _Float16* wrb_t = (_Float16*)carve((size_t)GP * HP * 2);
  _Float16* w1_t  = (_Float16*)carve((size_t)NP1 * DP * 2);
  float*    bpf   = (float*)carve((size_t)GP * 4);
  float*    bpb   = (float*)carve((size_t)GP * 4);
  float*    xkf   = (float*)carve((size_t)BS * GP * 4);
  float*    xkb   = (float*)carve((size_t)BS * GP * 4);
  float*    hidden= (float*)carve((size_t)BS * LDH * 4);
  _Float16* hid_h = (_Float16*)carve((size_t)BS * DP * 2);
  float*    outw1 = (float*)carve((size_t)BS * LDH * 4);
  float*    hhat  = (float*)carve((size_t)BS * LDH * 4);
  float*    c2buf = (float*)carve((size_t)BS * 4);
  (void)ws_size; (void)n_in; (void)in_sizes; (void)out_size;

  const int T = 256;
  // 1) pack weights (gate-permuted for LSTM path)
  pack_weight_t<<<(GP * DP + T - 1) / T, T, 0, stream>>>(Wk_f, DD, GG, wkf_t, DP, GP, 1);
  pack_weight_t<<<(GP * DP + T - 1) / T, T, 0, stream>>>(Wk_b, DD, GG, wkb_t, DP, GP, 1);
  pack_weight_t<<<(GP * HP + T - 1) / T, T, 0, stream>>>(Wr_f, HH, GG, wrf_t, HP, GP, 1);
  pack_weight_t<<<(GP * HP + T - 1) / T, T, 0, stream>>>(Wr_b, HH, GG, wrb_t, HP, GP, 1);
  pack_weight_t<<<(NP1 * DP + T - 1) / T, T, 0, stream>>>(W1, DD, DD, w1_t, DP, NP1, 0);
  pack_bias_perm<<<(GP + T - 1) / T, T, 0, stream>>>(b_f, bpf);
  pack_bias_perm<<<(GP + T - 1) / T, T, 0, stream>>>(b_b, bpb);

  // 2) embedding gather -> f16
  gather_pack_emb<<<(BS * DP + T - 1) / T, T, 0, stream>>>(batch_inputs, embm, emb_h);

  // 3) xk = emb @ Wk + b (permuted gate columns), both directions
  {
    int tiles = (BS / 16) * NTG;              // 512*38
    int blocks = (tiles + 7) / 8;             // 8 waves / block
    wmma_gemm_f16<<<blocks, T, 0, stream>>>(emb_h, wkf_t, bpf, GP, xkf, DP, GP, NTG, tiles);
    wmma_gemm_f16<<<blocks, T, 0, stream>>>(emb_h, wkb_t, bpb, GP, xkb, DP, GP, NTG, tiles);
  }

  // 4) sequential biLSTM scan: one persistent workgroup per direction.
  //    (215 KB dynamic LDS: legal on gfx1250's 320KB WGP; compile-only loop,
  //    so no hipFuncSetAttribute call is made here.)
  lstm_scan<<<2, 1024, LSTM_SMEM, stream>>>(xkf, xkb, wrf_t, wrb_t, hidden);

  // 5) out = hidden @ W1 + b1
  pack_rows_f16<<<(BS * DP + T - 1) / T, T, 0, stream>>>(hidden, LDH, DD, hid_h, DP, BS * DP);
  {
    int tiles = (BS / 16) * (NP1 / 16);       // 512*19
    int blocks = (tiles + 7) / 8;
    wmma_gemm_f16<<<blocks, T, 0, stream>>>(hid_h, w1_t, b1, DD, outw1, DP, LDH, NP1 / 16, tiles);
  }

  // 6) attention + final weighted sum over sequence
  attn_kernel<<<(BS + 7) / 8, T, 0, stream>>>(batch_inputs, embm, syn_idx, hidden,
                                              outw1, W2, b2, hhat, c2buf);
  reduce_kernel<<<BB, T, 0, stream>>>(hhat, c2buf, out);
}
